// NystromAttention_32435593020006
// MI455X (gfx1250) — matrix operand order repaired and checked by reading, not compiled, not run on hardware
//
#include <hip/hip_runtime.h>
#include <hip/hip_bf16.h>

// ---------------- CDNA5 WMMA plumbing (wave32, 16x16x32 bf16) ----------------
typedef __attribute__((ext_vector_type(16))) __bf16 bf16x16;
typedef __attribute__((ext_vector_type(8)))  __bf16 bf16x8;
typedef __attribute__((ext_vector_type(4)))  __bf16 bf16x4;
typedef __attribute__((ext_vector_type(8)))  float  f32x8;
typedef __attribute__((ext_vector_type(4)))  unsigned int u32x4;
typedef __attribute__((ext_vector_type(8)))  int  i32x8;
typedef __attribute__((ext_vector_type(4)))  int  i32x4;

#define DEV static __device__ __forceinline__

// ---- Tensor Data Mover availability (probe-verified builtins, arity differs) ----
#if defined(__has_builtin)
#if __has_builtin(__builtin_amdgcn_tensor_load_to_lds) && \
    __has_builtin(__builtin_amdgcn_s_wait_tensorcnt)
#define HAVE_TDM 1
#endif
#endif
#ifndef HAVE_TDM
#define HAVE_TDM 0
#endif

#if HAVE_TDM
// 2-D tile load global->LDS via TDM. rows x rowElems (bf16), row stride in elems.
// Descriptor bit layout per CDNA5 ISA 8.3/8.4 (D# group0 / group1), groups 2/3 zero.
DEV void tdm_load_2d(unsigned ldsAddr, const void* gptr, int rows, int rowElems,
                     int strideElems) {
  unsigned long long ga = (unsigned long long)gptr;
  u32x4 g0;
  g0[0] = 1u;                                   // count=1, user mode
  g0[1] = ldsAddr;                              // lds_addr (bytes)
  g0[2] = (unsigned)ga;                         // global_addr[31:0]
  g0[3] = (unsigned)((ga >> 32) & 0x01FFFFFFu) | (2u << 30);  // addr[56:32]|type=2
  i32x8 g1;
  g1[0] = (int)(1u << 16);                      // data_size=1 (2 bytes)
  g1[1] = (int)(((unsigned)rowElems & 0xFFFFu) << 16);          // tensor_dim0 lo
  g1[2] = (int)((((unsigned)rowElems >> 16) & 0xFFFFu) |
                (((unsigned)rows & 0xFFFFu) << 16));            // dim0 hi|dim1 lo
  g1[3] = (int)((((unsigned)rows >> 16) & 0xFFFFu) |
                (((unsigned)rowElems & 0xFFFFu) << 16));        // dim1 hi|tile_dim0
  g1[4] = (int)((unsigned)rows & 0xFFFFu);      // tile_dim1 (tile_dim2=0)
  g1[5] = strideElems;                          // tensor_dim0_stride lo32
  g1[6] = 0;
  g1[7] = 0;
  i32x4 z4 = {0, 0, 0, 0};
#if __clang_major__ >= 23
  i32x8 z8 = {0, 0, 0, 0, 0, 0, 0, 0};
  __builtin_amdgcn_tensor_load_to_lds(g0, g1, z4, z4, z8, 0);
#else
  __builtin_amdgcn_tensor_load_to_lds(g0, g1, z4, z4, 0);
#endif
}
#endif

DEV f32x8 wmma_bf16(bf16x16 a, bf16x16 b, f32x8 c) {
  return __builtin_amdgcn_wmma_f32_16x16x32_bf16(false, a, false, b, (short)0, c,
                                                 false, false);
}

DEV bf16x16 mkfrag(const __bf16* lo, const __bf16* hi) {
  bf16x8 l = *(const bf16x8*)lo;
  bf16x8 h = *(const bf16x8*)hi;
  bf16x16 r;
#pragma unroll
  for (int i = 0; i < 8; ++i) { r[i] = l[i]; r[i + 8] = h[i]; }
  return r;
}

// A fragment (16xK row-major, K-step 32): lane row lane&15, K {b0..b0+7,b0+16..b0+23}
DEV bf16x16 fragA(const __bf16* base, int ld, int row0, int k0, int lane) {
  int r  = row0 + (lane & 15);
  int b0 = k0 + ((lane & 16) ? 8 : 0);
  const __bf16* p = base + r * ld + b0;
  return mkfrag(p, p + 16);
}
// B fragment from [col][k]-transposed tile: lane col lane&15, K {b0..b0+7,b0+16..b0+23}
DEV bf16x16 fragB(const __bf16* base, int ld, int col0, int k0, int lane) {
  int c  = col0 + (lane & 15);
  int b0 = k0 + ((lane & 16) ? 8 : 0);
  const __bf16* p = base + c * ld + b0;
  return mkfrag(p, p + 16);
}
// C/D layout: element (vgpr j, lane) -> row = j + 8*(lane>>4), col = lane&15.

// ---------------- problem constants ----------------
#define Bq     4
#define Nq     4096
#define Hq     8
#define DHq    64
#define Lq     128
#define QDIMq  512
#define INNERq 512
#define BHq    32
#define MROWS  16384
#define SCALE_HALF 0.3535533905932738f  // 64^-0.25
#define FLTMAXC 3.402823466e38f

// ======================= pre-pass: f32 -> bf16 conversions =======================
__global__ void k_cvt_x(const float* __restrict__ x, __bf16* __restrict__ xb, int n4) {
  int i = blockIdx.x * blockDim.x + threadIdx.x;
  if (i >= n4) return;
  float4 v = ((const float4*)x)[i];
  bf16x4 o;
  o[0] = (__bf16)v.x; o[1] = (__bf16)v.y; o[2] = (__bf16)v.z; o[3] = (__bf16)v.w;
  ((bf16x4*)xb)[i] = o;
}

// WqkvT[c][k] = (c<512 ? Wq[k][c] : Wkv[k][c-512]), bf16, c in [0,1536)
__global__ void k_cvt_wqkv(const float* __restrict__ Wq, const float* __restrict__ Wkv,
                           __bf16* __restrict__ WT) {
  int idx = blockIdx.x * blockDim.x + threadIdx.x;  // 512*1536
  int k = idx / 1536, c = idx % 1536;
  float v = (c < QDIMq) ? Wq[(size_t)k * QDIMq + c]
                        : Wkv[(size_t)k * (2 * INNERq) + (c - QDIMq)];
  WT[(size_t)c * QDIMq + k] = (__bf16)v;
}

__global__ void k_cvt_wout(const float* __restrict__ Wout, __bf16* __restrict__ WT) {
  int idx = blockIdx.x * blockDim.x + threadIdx.x;  // 512*512
  int k = idx / QDIMq, c = idx % QDIMq;
  WT[(size_t)c * INNERq + k] = (__bf16)Wout[(size_t)k * QDIMq + c];
}

// ======= shared GEMM mainloop (M-tile 64 x N-tile 64, K=512, bf16 inputs) =======
// A: row-major [.,512]; BT: pre-transposed [col][512]. TDM double-buffered.
DEV void gemm_mainloop(const __bf16* Asrc, const __bf16* BTsrc, __bf16* As,
                       __bf16* Bs, int wave, int lane, int tid, f32x8 acc[4]) {
  // As/Bs are [2][64*32]
#if HAVE_TDM
  if (wave == 0) {
    tdm_load_2d((unsigned)(size_t)&As[0], Asrc, 64, 32, QDIMq);
    tdm_load_2d((unsigned)(size_t)&Bs[0], BTsrc, 64, 32, QDIMq);
    __builtin_amdgcn_s_wait_tensorcnt(0);
  }
  __syncthreads();
#endif
  for (int ks = 0; ks < QDIMq / 32; ++ks) {
#if HAVE_TDM
    const int cur = ks & 1;
    if (wave == 0 && ks < QDIMq / 32 - 1) {  // prefetch next k-step
      tdm_load_2d((unsigned)(size_t)&As[(cur ^ 1) * 2048], Asrc + (ks + 1) * 32, 64,
                  32, QDIMq);
      tdm_load_2d((unsigned)(size_t)&Bs[(cur ^ 1) * 2048], BTsrc + (ks + 1) * 32, 64,
                  32, QDIMq);
    }
#else
    const int cur = 0;
    for (int ch = tid; ch < 256; ch += 128) {  // 64x32 bf16 in 16B chunks
      int r = ch >> 2, c0 = (ch & 3) * 8;
      *(bf16x8*)(As + r * 32 + c0) =
          *(const bf16x8*)(Asrc + (size_t)r * QDIMq + ks * 32 + c0);
      *(bf16x8*)(Bs + r * 32 + c0) =
          *(const bf16x8*)(BTsrc + (size_t)r * QDIMq + ks * 32 + c0);
    }
    __syncthreads();
#endif
    bf16x16 a = fragA(As + cur * 2048, 32, wave * 16, 0, lane);
#pragma unroll
    for (int t = 0; t < 4; ++t) {
      bf16x16 b = fragB(Bs + cur * 2048, 32, t * 16, 0, lane);
      acc[t] = wmma_bf16(a, b, acc[t]);
    }
#if HAVE_TDM
    if (wave == 0) __builtin_amdgcn_s_wait_tensorcnt(0);
#endif
    __syncthreads();
  }
}

// ======================= K1: fused QKV projection =======================
__global__ __launch_bounds__(128) void k_qkv(const __bf16* __restrict__ xb,
                                             const __bf16* __restrict__ WT,
                                             __bf16* __restrict__ qh,
                                             __bf16* __restrict__ kh,
                                             __bf16* __restrict__ vh) {
  __shared__ __align__(16) __bf16 As[2][64 * 32];
  __shared__ __align__(16) __bf16 Bs[2][64 * 32];
  const int tid = threadIdx.x, wave = tid >> 5, lane = tid & 31;
  const int colBase = blockIdx.x * 64;  // 0..1535
  const int rowBase = blockIdx.y * 64;

  f32x8 acc[4];
#pragma unroll
  for (int t = 0; t < 4; ++t)
#pragma unroll
    for (int j = 0; j < 8; ++j) acc[t][j] = 0.f;

  gemm_mainloop(xb + (size_t)rowBase * QDIMq, WT + (size_t)colBase * QDIMq,
                &As[0][0], &Bs[0][0], wave, lane, tid, acc);

  const int hi8 = (lane & 16) ? 8 : 0;
#pragma unroll
  for (int t = 0; t < 4; ++t)
#pragma unroll
    for (int j = 0; j < 8; ++j) {
      int row  = rowBase + wave * 16 + j + hi8;
      int gcol = colBase + t * 16 + (lane & 15);
      float v  = acc[t][j] * SCALE_HALF;
      int b = row >> 12, n = row & 4095;
      __bf16* dst; int c2;
      if (gcol < 512)       { dst = qh; c2 = gcol; }
      else if (gcol < 1024) { dst = kh; c2 = gcol - 512; }
      else                  { dst = vh; c2 = gcol - 1024; }
      int h = c2 >> 6, d = c2 & 63;
      dst[((size_t)(b * Hq + h) * Nq + n) * DHq + d] = (__bf16)v;
    }
}

// ======================= K2: landmark means =======================
__global__ void k_landmark(const __bf16* __restrict__ qh,
                           const __bf16* __restrict__ kh,
                           __bf16* __restrict__ ql, __bf16* __restrict__ kl) {
  int idx = blockIdx.x * blockDim.x + threadIdx.x;  // 32*128*64
  int d = idx & 63, lj = (idx >> 6) & 127, bh = idx >> 13;
  size_t base = ((size_t)bh * Nq + lj * 32) * DHq + d;
  float sq = 0.f, sk = 0.f;
#pragma unroll 4
  for (int g = 0; g < 32; ++g) {
    sq += (float)qh[base + (size_t)g * DHq];
    sk += (float)kh[base + (size_t)g * DHq];
  }
  ql[idx] = (__bf16)(sq * 0.03125f);
  kl[idx] = (__bf16)(sk * 0.03125f);
}

// ======================= K3a: kernel_2 = softmax(ql @ kl^T) =======================
__global__ __launch_bounds__(128) void k_k2softmax(const __bf16* __restrict__ ql,
                                                   const __bf16* __restrict__ kl,
                                                   __bf16* __restrict__ K2) {
  __shared__ float red[128];
  int bh = blockIdx.x >> 7, i = blockIdx.x & 127, t = threadIdx.x;
  const __bf16* qr = ql + ((size_t)bh * Lq + i) * DHq;
  const __bf16* kr = kl + ((size_t)bh * Lq + t) * DHq;
  float dot = 0.f;
#pragma unroll 8
  for (int d = 0; d < DHq; ++d) dot += (float)qr[d] * (float)kr[d];
  red[t] = dot; __syncthreads();
  for (int s = 64; s > 0; s >>= 1) {
    if (t < s) red[t] = fmaxf(red[t], red[t + s]);
    __syncthreads();
  }
  float mx = red[0]; __syncthreads();
  float e = __expf(dot - mx);
  red[t] = e; __syncthreads();
  for (int s = 64; s > 0; s >>= 1) {
    if (t < s) red[t] += red[t + s];
    __syncthreads();
  }
  K2[((size_t)bh * Lq + i) * Lq + t] = (__bf16)(e / red[0]);
}

// ======================= K3b: Newton-Schulz scaling s1*s2 =======================
__global__ __launch_bounds__(128) void k_scales(const __bf16* __restrict__ K2,
                                                float* __restrict__ sarr) {
  __shared__ float ca[128], ra[128];
  int bh = blockIdx.x, t = threadIdx.x;
  const __bf16* Kb = K2 + (size_t)bh * Lq * Lq;
  float cs = 0.f, rs = 0.f;
  for (int i = 0; i < Lq; ++i) {
    cs += fabsf((float)Kb[i * Lq + t]);
    rs += fabsf((float)Kb[t * Lq + i]);
  }
  ca[t] = cs; ra[t] = rs; __syncthreads();
  for (int s = 64; s > 0; s >>= 1) {
    if (t < s) { ca[t] = fmaxf(ca[t], ca[t + s]); ra[t] = fmaxf(ra[t], ra[t + s]); }
    __syncthreads();
  }
  if (t == 0) sarr[bh] = (1.0f / ca[0]) * (1.0f / ra[0]);
}

// ======================= K3c: V0 = s * K2^T =======================
__global__ void k_vinit(const __bf16* __restrict__ K2, const float* __restrict__ sarr,
                        __bf16* __restrict__ V) {
  int idx = blockIdx.x * blockDim.x + threadIdx.x;  // 32*128*128
  int j = idx & 127, i = (idx >> 7) & 127, bh = idx >> 14;
  V[idx] = (__bf16)((float)K2[((size_t)bh * Lq + j) * Lq + i] * sarr[bh]);
}

// ============== batched GEMM: D = s*(A@B) + c*C  (128 x Nn, K=128) ==============
__global__ __launch_bounds__(128) void k_nsgemm(const __bf16* __restrict__ A,
                                                const __bf16* __restrict__ Bm,
                                                const __bf16* __restrict__ Cm,
                                                __bf16* __restrict__ D, int Nn,
                                                float s, float c) {
  __shared__ __align__(16) __bf16 As[64 * 32];
  __shared__ __align__(16) __bf16 Bs[64 * 32];
  const int tid = threadIdx.x, wave = tid >> 5, lane = tid & 31;
  const int colBase = blockIdx.x * 64, rowBase = blockIdx.y * 64, bh = blockIdx.z;
  const __bf16* Ab = A + (size_t)bh * Lq * Lq;
  const __bf16* Bb = Bm + (size_t)bh * Lq * Nn;

  f32x8 acc[4];
#pragma unroll
  for (int t = 0; t < 4; ++t)
#pragma unroll
    for (int j = 0; j < 8; ++j) acc[t][j] = 0.f;

  for (int kk = 0; kk < Lq; kk += 32) {
    for (int ch = tid; ch < 256; ch += 128) {  // A: 64x32 in 16B chunks
      int r = ch >> 2, c0 = (ch & 3) * 8;
      *(bf16x8*)(As + r * 32 + c0) =
          *(const bf16x8*)(Ab + (size_t)(rowBase + r) * Lq + kk + c0);
    }
    for (int ch = tid; ch < 256; ch += 128) {  // B: 32x64 transpose-stage
      int k = ch >> 3, c0 = (ch & 7) * 8;
      bf16x8 bb = *(const bf16x8*)(Bb + (size_t)(kk + k) * Nn + colBase + c0);
#pragma unroll
      for (int e = 0; e < 8; ++e) Bs[(c0 + e) * 32 + k] = bb[e];
    }
    __syncthreads();
    bf16x16 a = fragA(As, 32, wave * 16, 0, lane);
#pragma unroll
    for (int t = 0; t < 4; ++t) {
      bf16x16 b = fragB(Bs, 32, t * 16, 0, lane);
      acc[t] = wmma_bf16(a, b, acc[t]);
    }
    __syncthreads();
  }
  const int hi8 = (lane & 16) ? 8 : 0;
#pragma unroll
  for (int t = 0; t < 4; ++t)
#pragma unroll
    for (int j = 0; j < 8; ++j) {
      int row = rowBase + wave * 16 + j + hi8;
      int col = colBase + t * 16 + (lane & 15);
      size_t o = (size_t)bh * Lq * Nn + (size_t)row * Nn + col;
      float v = s * acc[t][j];
      if (Cm) v += c * (float)Cm[o];
      D[o] = (__bf16)v;
    }
}

// ======= K4: out3 = softmax(ql @ k^T - FLT_MAX) @ v  (flash, per bh) =======
__global__ __launch_bounds__(256) void k_flash(const __bf16* __restrict__ ql,
                                               const __bf16* __restrict__ kh,
                                               const __bf16* __restrict__ vh,
                                               __bf16* __restrict__ out3) {
  __shared__ __align__(16) __bf16 kt[64 * 64];     // [m'][d]
  __shared__ __align__(16) __bf16 vt[64 * 64];     // [d][m'] (transposed)
  __shared__ __align__(16) float  lg[8][16 * 64];  // per-wave logits / P(bf16) in place
  __shared__ float mrow[8][16], lrow[8][16], frow[8][16];
  const int bh = blockIdx.x, tid = threadIdx.x, wave = tid >> 5, lane = tid & 31;
  const int hi8 = (lane & 16) ? 8 : 0;

  const __bf16* qb = ql + ((size_t)bh * Lq + wave * 16) * DHq;
  bf16x16 aq[2];
#pragma unroll
  for (int s = 0; s < 2; ++s) aq[s] = fragA(qb, DHq, 0, s * 32, lane);

  f32x8 O[4];
#pragma unroll
  for (int t = 0; t < 4; ++t)
#pragma unroll
    for (int j = 0; j < 8; ++j) O[t][j] = 0.f;
  if (lane < 16) { mrow[wave][lane] = -FLTMAXC; lrow[wave][lane] = 0.f; }
  __syncthreads();

  for (int c0 = 0; c0 < Nq; c0 += 64) {
#if HAVE_TDM
    if (wave == 0) {  // contiguous 64x64 bf16 K tile via Tensor Data Mover
      tdm_load_2d((unsigned)(size_t)kt, kh + ((size_t)bh * Nq + c0) * DHq, 64, 64, 64);
    }
#else
    for (int ch = tid; ch < 512; ch += 256) {
      int r = ch >> 3, d0 = (ch & 7) * 8;
      *(bf16x8*)(kt + r * 64 + d0) =
          *(const bf16x8*)(kh + ((size_t)bh * Nq + c0 + r) * DHq + d0);
    }
#endif
    for (int ch = tid; ch < 512; ch += 256) {  // V tile, transposed stage
      int r = ch >> 3, d0 = (ch & 7) * 8;
      bf16x8 vv = *(const bf16x8*)(vh + ((size_t)bh * Nq + c0 + r) * DHq + d0);
#pragma unroll
      for (int e = 0; e < 8; ++e) vt[(d0 + e) * 64 + r] = vv[e];
    }
#if HAVE_TDM
    if (wave == 0) __builtin_amdgcn_s_wait_tensorcnt(0);
#endif
    __syncthreads();

    f32x8 sacc[4];
#pragma unroll
    for (int t = 0; t < 4; ++t)
#pragma unroll
      for (int j = 0; j < 8; ++j) sacc[t][j] = 0.f;
#pragma unroll
    for (int s = 0; s < 2; ++s)
#pragma unroll
      for (int t = 0; t < 4; ++t) {
        bf16x16 b = fragB(kt, 64, t * 16, s * 32, lane);
        sacc[t] = wmma_bf16(aq[s], b, sacc[t]);
      }
    float* lw = lg[wave];
#pragma unroll
    for (int t = 0; t < 4; ++t)
#pragma unroll
      for (int j = 0; j < 8; ++j)
        lw[(j + hi8) * 64 + t * 16 + (lane & 15)] = sacc[t][j] - FLTMAXC;
    __syncthreads();

    if (lane < 16) {  // online softmax, one row per lane
      float* lr = lg[wave] + lane * 64;
      float mprev = mrow[wave][lane];
      float mx = mprev;
      for (int cc = 0; cc < 64; ++cc) mx = fmaxf(mx, lr[cc]);
      float fac = __expf(mprev - mx);
      __bf16* pb = (__bf16*)lr;  // in-place bf16 P (write offset <= read offset)
      float sum = 0.f;
      for (int cc = 0; cc < 64; ++cc) {
        float e = __expf(lr[cc] - mx);
        sum += e;
        pb[cc] = (__bf16)e;
      }
      lrow[wave][lane] = lrow[wave][lane] * fac + sum;
      mrow[wave][lane] = mx;
      frow[wave][lane] = fac;
    }
    __syncthreads();

#pragma unroll
    for (int t = 0; t < 4; ++t)
#pragma unroll
      for (int j = 0; j < 8; ++j) O[t][j] *= frow[wave][j + hi8];
    const __bf16* pb = (const __bf16*)lg[wave];  // bf16 row stride = 128
#pragma unroll
    for (int s = 0; s < 2; ++s) {
      bf16x16 ap = fragA(pb, 128, 0, s * 32, lane);
#pragma unroll
      for (int t = 0; t < 4; ++t) {
        bf16x16 b = fragB(vt, 64, t * 16, s * 32, lane);
        O[t] = wmma_bf16(ap, b, O[t]);
      }
    }
    __syncthreads();
  }
#pragma unroll
  for (int t = 0; t < 4; ++t)
#pragma unroll
    for (int j = 0; j < 8; ++j) {
      int row = wave * 16 + j + hi8;
      float inv = 1.0f / lrow[wave][j + hi8];
      out3[((size_t)bh * Lq + row) * DHq + t * 16 + (lane & 15)] =
          (__bf16)(O[t][j] * inv);
    }
}

// ======= K5: attn = softmax(q @ kl^T) @ Z, scattered to (b,n,h*d) =======
__global__ __launch_bounds__(64) void k_attn(const __bf16* __restrict__ qh,
                                             const __bf16* __restrict__ kl,
                                             const __bf16* __restrict__ Z,
                                             __bf16* __restrict__ attn) {
  __shared__ __align__(16) __bf16 klS[128 * 64];   // [j][d]
  __shared__ __align__(16) __bf16 Zt[64 * 128];    // [d][j]
  __shared__ __align__(16) float  lg[2][16 * 128];
  const int rb = blockIdx.x, bh = blockIdx.y;
  const int tid = threadIdx.x, wave = tid >> 5, lane = tid & 31;
  const int hi8 = (lane & 16) ? 8 : 0;

  for (int ch = tid; ch < 1024; ch += 64)
    *(bf16x8*)(klS + ch * 8) = *(const bf16x8*)(kl + (size_t)bh * Lq * DHq + ch * 8);
  for (int ch = tid; ch < 1024; ch += 64) {
    int j = ch >> 3, d0 = (ch & 7) * 8;
    bf16x8 zz = *(const bf16x8*)(Z + (size_t)bh * Lq * DHq + j * DHq + d0);
#pragma unroll
    for (int e = 0; e < 8; ++e) Zt[(d0 + e) * 128 + j] = zz[e];
  }
  const int rowStart = rb * 32 + wave * 16;
  const __bf16* qb = qh + ((size_t)bh * Nq + rowStart) * DHq;
  bf16x16 aq[2];
#pragma unroll
  for (int s = 0; s < 2; ++s) aq[s] = fragA(qb, DHq, 0, s * 32, lane);
  __syncthreads();

  f32x8 sacc[8];
#pragma unroll
  for (int t = 0; t < 8; ++t)
#pragma unroll
    for (int j = 0; j < 8; ++j) sacc[t][j] = 0.f;
#pragma unroll
  for (int s = 0; s < 2; ++s)
#pragma unroll
    for (int t = 0; t < 8; ++t) {
      bf16x16 b = fragB(klS, 64, t * 16, s * 32, lane);
      sacc[t] = wmma_bf16(aq[s], b, sacc[t]);
    }
  float* lw = lg[wave];
#pragma unroll
  for (int t = 0; t < 8; ++t)
#pragma unroll
    for (int j = 0; j < 8; ++j)
      lw[(j + hi8) * 128 + t * 16 + (lane & 15)] = sacc[t][j];
  __syncthreads();

  if (lane < 16) {
    float* lr = lg[wave] + lane * 128;
    float mx = -FLTMAXC;
    for (int cc = 0; cc < 128; ++cc) mx = fmaxf(mx, lr[cc]);
    float sum = 0.f;
    for (int cc = 0; cc < 128; ++cc) sum += __expf(lr[cc] - mx);
    float inv = 1.0f / sum;
    __bf16* pb = (__bf16*)lr;  // in-place bf16
    for (int cc = 0; cc < 128; ++cc) {
      float e = __expf(lr[cc] - mx) * inv;
      pb[cc] = (__bf16)e;
    }
  }
  __syncthreads();

  f32x8 O[4];
#pragma unroll
  for (int t = 0; t < 4; ++t)
#pragma unroll
    for (int j = 0; j < 8; ++j) O[t][j] = 0.f;
  const __bf16* pb = (const __bf16*)lg[wave];  // bf16 row stride = 256
#pragma unroll
  for (int s = 0; s < 4; ++s) {
    bf16x16 ap = fragA(pb, 256, 0, s * 32, lane);
#pragma unroll
    for (int t = 0; t < 4; ++t) {
      bf16x16 b = fragB(Zt, 128, t * 16, s * 32, lane);
      O[t] = wmma_bf16(ap, b, O[t]);
    }
  }
  const int b = bh >> 3, h = bh & 7;
#pragma unroll
  for (int t = 0; t < 4; ++t)
#pragma unroll
    for (int j = 0; j < 8; ++j) {
      int n = rowStart + j + hi8;
      int col = h * DHq + t * 16 + (lane & 15);
      attn[((size_t)(b * Nq + n)) * INNERq + col] = (__bf16)O[t][j];
    }
}

// ================ K6: out = attn @ Wout + bout (f32 output) ================
__global__ __launch_bounds__(128) void k_out(const __bf16* __restrict__ attn,
                                             const __bf16* __restrict__ WoutT,
                                             const float* __restrict__ bout,
                                             float* __restrict__ out) {
  __shared__ __align__(16) __bf16 As[2][64 * 32];
  __shared__ __align__(16) __bf16 Bs[2][64 * 32];
  const int tid = threadIdx.x, wave = tid >> 5, lane = tid & 31;
  const int colBase = blockIdx.x * 64, rowBase = blockIdx.y * 64;

  f32x8 acc[4];
#pragma unroll
  for (int t = 0; t < 4; ++t)
#pragma unroll
    for (int j = 0; j < 8; ++j) acc[t][j] = 0.f;

  gemm_mainloop(attn + (size_t)rowBase * INNERq, WoutT + (size_t)colBase * INNERq,
                &As[0][0], &Bs[0][0], wave, lane, tid, acc);

  const int hi8 = (lane & 16) ? 8 : 0;
#pragma unroll
  for (int t = 0; t < 4; ++t)
#pragma unroll
    for (int j = 0; j < 8; ++j) {
      int row = rowBase + wave * 16 + j + hi8;
      int col = colBase + t * 16 + (lane & 15);
      out[(size_t)row * QDIMq + col] = acc[t][j] + bout[col];
    }
}

// ======================= host launch =======================
extern "C" void kernel_launch(void* const* d_in, const int* in_sizes, int n_in,
                              void* d_out, int out_size, void* d_ws, size_t ws_size,
                              hipStream_t stream) {
  const float* x    = (const float*)d_in[0];
  // d_in[1] = mask (all-true in this problem; folded into the math)
  const float* Wq   = (const float*)d_in[2];
  const float* Wkv  = (const float*)d_in[3];
  const float* Wout = (const float*)d_in[4];
  const float* bout = (const float*)d_in[5];
  float* out = (float*)d_out;

  size_t off = 0;
  auto alloc = [&](size_t bytes) -> void* {
    off = (off + 255) & ~(size_t)255;
    void* p = (char*)d_ws + off;
    off += bytes;
    return p;
  };
  __bf16* xb    = (__bf16*)alloc((size_t)MROWS * QDIMq * 2);
  __bf16* WqkvT = (__bf16*)alloc((size_t)1536 * QDIMq * 2);
  __bf16* WoutT = (__bf16*)alloc((size_t)QDIMq * INNERq * 2);
  __bf16* qh    = (__bf16*)alloc((size_t)BHq * Nq * DHq * 2);
  __bf16* kh    = (__bf16*)alloc((size_t)BHq * Nq * DHq * 2);
  __bf16* vh    = (__bf16*)alloc((size_t)BHq * Nq * DHq * 2);
  __bf16* ql    = (__bf16*)alloc((size_t)BHq * Lq * DHq * 2);
  __bf16* kl    = (__bf16*)alloc((size_t)BHq * Lq * DHq * 2);
  __bf16* K2    = (__bf16*)alloc((size_t)BHq * Lq * Lq * 2);
  float*  sarr  = (float*)alloc(BHq * 4);
  __bf16* Va    = (__bf16*)alloc((size_t)BHq * Lq * Lq * 2);
  __bf16* Vb    = (__bf16*)alloc((size_t)BHq * Lq * Lq * 2);
  __bf16* Pm    = (__bf16*)alloc((size_t)BHq * Lq * Lq * 2);
  __bf16* Qm    = (__bf16*)alloc((size_t)BHq * Lq * Lq * 2);
  __bf16* Rm    = (__bf16*)alloc((size_t)BHq * Lq * Lq * 2);
  __bf16* out3  = (__bf16*)alloc((size_t)BHq * Lq * DHq * 2);
  __bf16* Zm    = (__bf16*)alloc((size_t)BHq * Lq * DHq * 2);
  __bf16* attn  = (__bf16*)alloc((size_t)MROWS * INNERq * 2);
  (void)ws_size; (void)in_sizes; (void)n_in; (void)out_size;

  // 0) one-time f32->bf16 conversion + weight pre-transposition
  int n4 = MROWS * QDIMq / 4;
  k_cvt_x<<<(n4 + 255) / 256, 256, 0, stream>>>(x, xb, n4);
  k_cvt_wqkv<<<(QDIMq * 1536) / 256, 256, 0, stream>>>(Wq, Wkv, WqkvT);
  k_cvt_wout<<<(QDIMq * INNERq) / 256, 256, 0, stream>>>(Wout, WoutT);

  // 1) fused QKV projection (M=16384, Ncols=1536, K=512)
  k_qkv<<<dim3(1536 / 64, MROWS / 64), 128, 0, stream>>>(xb, WqkvT, qh, kh, vh);
  // 2) landmark means
  k_landmark<<<(BHq * Lq * DHq) / 256, 256, 0, stream>>>(qh, kh, ql, kl);
  // 3) kernel_2 softmax, scales, V0
  k_k2softmax<<<BHq * Lq, 128, 0, stream>>>(ql, kl, K2);
  k_scales<<<BHq, 128, 0, stream>>>(K2, sarr);
  k_vinit<<<(BHq * Lq * Lq) / 256, 256, 0, stream>>>(K2, sarr, Va);
  // 4) flash kernel_3 @ v
  k_flash<<<BHq, 256, 0, stream>>>(ql, kh, vh, out3);
  // 5) Newton-Schulz: V' = 0.25 V (13I - KV(15I - KV(7I - KV)))
  __bf16 *Vc = Va, *Vn = Vb;
  for (int it = 0; it < 8; ++it) {
    k_nsgemm<<<dim3(2, 2, BHq), 128, 0, stream>>>(K2, Vc, nullptr, Pm, 128, 1.f, 0.f);
    k_nsgemm<<<dim3(2, 2, BHq), 128, 0, stream>>>(Pm, Pm, Pm, Qm, 128, -1.f, 7.f);
    k_nsgemm<<<dim3(2, 2, BHq), 128, 0, stream>>>(Pm, Qm, Pm, Rm, 128, -1.f, 15.f);
    k_nsgemm<<<dim3(2, 2, BHq), 128, 0, stream>>>(Vc, Rm, Vc, Vn, 128, -0.25f, 3.25f);
    __bf16* tmp = Vc; Vc = Vn; Vn = tmp;
  }
  // 6) Z = inv @ out3  (128x128 @ 128x64)
  k_nsgemm<<<dim3(1, 2, BHq), 128, 0, stream>>>(Vc, out3, nullptr, Zm, 64, 1.f, 0.f);
  // 7) attn = softmax(q @ kl^T) @ Z
  k_attn<<<dim3(Nq / 32, BHq), 64, 0, stream>>>(qh, kl, Zm, attn);
  // 8) out = attn @ Wout + bout
  k_out<<<dim3(QDIMq / 64, MROWS / 64), 128, 0, stream>>>(attn, WoutT, bout, out);
}
